// ETLayer_2886218023209
// MI455X (gfx1250) — matrix operand order repaired and checked by reading, not compiled, not run on hardware
//
#include <hip/hip_runtime.h>
#include <hip/hip_bf16.h>
#include <math.h>

typedef __attribute__((ext_vector_type(16))) __bf16 v16bf;
typedef __attribute__((ext_vector_type(8)))  __bf16 v8bf;
typedef __attribute__((ext_vector_type(8)))  float  v8f;

#define BN 4096
#define T  64
#define H  128

// ---------------------------------------------------------------------------
// WMMA fragment helpers (CDNA5 wave32 16x16x32 bf16; this code's contiguous K order, same for A and B)
// A: [M,K] row-major. lane l: M=l&15,
//                     elements idx=0..15 are K = (l>>4)*16 + idx
// B supplied as Bt [N,K] row-major (K contiguous): lane l: N=l&15,
//                     elements idx=0..15 are K = (l>>4)*16 + idx
// C/D f32: lane l: N=l&15, c[r] = C[M = r + 8*(l>>4)][N]
// ---------------------------------------------------------------------------
__device__ __forceinline__ v16bf load_A(const __bf16* __restrict__ A, int ldA,
                                        int m0, int k0, int lane) {
  int m  = m0 + (lane & 15);
  int kb = k0 + ((lane >> 4) << 4);
  const __bf16* p = A + m * ldA + kb;
  return *(const v16bf*)(p);          // K offsets +0..15 (32B aligned)
}

__device__ __forceinline__ v16bf load_B(const __bf16* __restrict__ Bt, int ldB,
                                        int n0, int k0, int lane) {
  int n  = n0 + (lane & 15);
  int kb = k0 + ((lane >> 4) << 4);
  return *(const v16bf*)(Bt + n * ldB + kb);   // 32B aligned
}

__device__ __forceinline__ void gemm_acc(v8f& c, const __bf16* __restrict__ A, int ldA,
                                         const __bf16* __restrict__ Bt, int ldB,
                                         int m0, int n0, int K, int lane) {
#pragma unroll
  for (int k0 = 0; k0 < K; k0 += 32) {
    v16bf a = load_A(A, ldA, m0, k0, lane);
    v16bf b = load_B(Bt, ldB, n0, k0, lane);
    c = __builtin_amdgcn_wmma_f32_16x16x32_bf16(false, a, false, b,
                                                (short)0, c, false, false);
  }
}

// OUT[64,128] = act(A[64,128] @ Wt^T + bias), bf16 out (+optional transposed copy)
template<bool SILU, bool TRANSPOSE>
__device__ __forceinline__ void dense_64x128(const __bf16* __restrict__ A, int ldA,
                                             const __bf16* __restrict__ Wt,
                                             const float* __restrict__ bias,
                                             __bf16* __restrict__ out,
                                             __bf16* __restrict__ outT, int tid) {
  int lane = tid & 31, wv = tid >> 5;
  int mt = (wv & 3) << 4;
  int mb = mt + ((lane >> 4) << 3);
  int nb = (wv >> 2) << 6;
#pragma unroll
  for (int i = 0; i < 4; ++i) {
    int nt = nb + (i << 4);
    v8f c = {};
    gemm_acc(c, A, ldA, Wt, 128, mt, nt, 128, lane);
    int n = nt + (lane & 15);
    float bv = bias[n];
#pragma unroll
    for (int r = 0; r < 8; ++r) {
      float xv = c[r] + bv;
      if (SILU) xv = xv / (1.0f + __expf(-xv));
      out[(mb + r) * 128 + n] = (__bf16)xv;
      if (TRANSPOSE) outT[n * 64 + (mb + r)] = (__bf16)xv;
    }
  }
}

// ---------------------------------------------------------------------------
// Shared-memory layout (typed object so LDS address-space inference is exact)
// ---------------------------------------------------------------------------
struct __align__(32) Smem {
  float  x_s[192];          // x[3][64]
  float  vx[64];
  float  rowsum[64];
  __bf16 q_s[64 * 128];     // [t][h]
  __bf16 k_s[64 * 128];     // [s][h]
  __bf16 v_s[64 * 128];     // [s][h]
  __bf16 vT_s[128 * 64];    // [h][s]
  __bf16 alpha_s[64 * 64];  // [t][s]
  __bf16 aSh_s[64 * 128];   // [t][r]
  float  sc[64 * 64];       // scores -> w
  union Scr {
    struct TokHid { __bf16 tok[64 * 128]; __bf16 hid[64 * 128]; } th;
    float P[64 * 128];      // [t][j], reuses tok+hid after MLPs
  } scr;
};

// ---------------------------------------------------------------------------
// Prep kernel 1: bf16 transposed weights Wt[o][i] = W[i][o] for 7 matrices
// ---------------------------------------------------------------------------
__global__ void prep_weights(const float* __restrict__ qw1, const float* __restrict__ qw2,
                             const float* __restrict__ kw1, const float* __restrict__ kw2,
                             const float* __restrict__ vw1, const float* __restrict__ vw2,
                             const float* __restrict__ xmw1, __bf16* __restrict__ dst) {
  int g = blockIdx.x * blockDim.x + threadIdx.x;
  if (g >= 7 * 16384) return;
  int mat = g >> 14, rem = g & 16383;
  int o = rem >> 7, i = rem & 127;
  const float* src[7] = {qw1, qw2, kw1, kw2, vw1, vw2, xmw1};
  dst[g] = (__bf16)src[mat][i * 128 + o];
}

// ---------------------------------------------------------------------------
// Prep kernel 2: R[j] = emb(j-63) @ t_w + t_b  for j=0..126 (row 127 = 0).
// Rbf: [j][h] (K-contiguous B for q@R^T); Rt: [h][j] (K-contiguous B for aShift@R)
// ---------------------------------------------------------------------------
__global__ void prep_R(const float* __restrict__ t_w, const float* __restrict__ t_b,
                       __bf16* __restrict__ Rbf, __bf16* __restrict__ Rt) {
  __shared__ float emb[128];
  int j = blockIdx.x;          // 0..127
  int hh = threadIdx.x;        // 0..127
  if (j == 127) {
    Rbf[j * 128 + hh] = (__bf16)0.0f;
    Rt[hh * 128 + j]  = (__bf16)0.0f;
    return;
  }
  float rel = (float)(j - 63);
  if (hh < 64) {
    float f = __expf((float)hh * (-logf(10000.0f) / 63.0f));
    float arg = rel * f;
    emb[hh]      = sinf(arg);
    emb[hh + 64] = cosf(arg);
  }
  __syncthreads();
  float acc = t_b[hh];
  for (int e = 0; e < 128; ++e) acc += emb[e] * t_w[e * 128 + hh];
  Rbf[j * 128 + hh] = (__bf16)acc;
  Rt[hh * 128 + j]  = (__bf16)acc;
}

// ---------------------------------------------------------------------------
// Main fused per-node kernel: one block per node, 256 threads (8 wave32)
// ---------------------------------------------------------------------------
__global__ __launch_bounds__(256) void et_layer_main(
    const float* __restrict__ x, const float* __restrict__ h,
    const float* __restrict__ q_b1, const float* __restrict__ q_b2,
    const float* __restrict__ k_b1, const float* __restrict__ k_b2,
    const float* __restrict__ v_b1, const float* __restrict__ v_b2,
    const float* __restrict__ xm_b1, const float* __restrict__ xm_w2,
    const float* __restrict__ xm_b2,
    const __bf16* __restrict__ Wq1, const __bf16* __restrict__ Wq2,
    const __bf16* __restrict__ Wk1, const __bf16* __restrict__ Wk2,
    const __bf16* __restrict__ Wv1, const __bf16* __restrict__ Wv2,
    const __bf16* __restrict__ Wxm1,
    const __bf16* __restrict__ Rbf, const __bf16* __restrict__ Rt,
    float* __restrict__ x_out, float* __restrict__ h_out) {
  __shared__ Smem sm;

  int n   = blockIdx.x;
  int tid = threadIdx.x;
  int lane = tid & 31, wv = tid >> 5;
  const float* hN = h + (size_t)n * H * T;
  const float* xN = x + (size_t)n * 3 * T;

  // ---- load x and transpose h -> tok (bf16) ----
  if (tid < 192) sm.x_s[tid] = xN[tid];
  for (int idx = tid; idx < H * T; idx += 256) {
    int t = idx & 63, cch = idx >> 6;
    sm.scr.th.tok[t * 128 + cch] = (__bf16)hN[cch * 64 + t];
  }
  __syncthreads();

  // ---- q / k / v MLPs + xm hidden (all WMMA) ----
  dense_64x128<true,  false>(sm.scr.th.tok, 128, Wq1, q_b1, sm.scr.th.hid, nullptr, tid);
  __syncthreads();
  dense_64x128<false, false>(sm.scr.th.hid, 128, Wq2, q_b2, sm.q_s, nullptr, tid);
  __syncthreads();
  dense_64x128<true,  false>(sm.scr.th.tok, 128, Wk1, k_b1, sm.scr.th.hid, nullptr, tid);
  __syncthreads();
  dense_64x128<false, false>(sm.scr.th.hid, 128, Wk2, k_b2, sm.k_s, nullptr, tid);
  __syncthreads();
  dense_64x128<true,  false>(sm.scr.th.tok, 128, Wv1, v_b1, sm.scr.th.hid, nullptr, tid);
  __syncthreads();
  dense_64x128<false, true >(sm.scr.th.hid, 128, Wv2, v_b2, sm.v_s, sm.vT_s, tid);
  __syncthreads();
  dense_64x128<true,  false>(sm.v_s, 128, Wxm1, xm_b1, sm.scr.th.hid, nullptr, tid);
  __syncthreads();

  // ---- vx[t] = hid[t] . xm_w2 + b (matvec, VALU) ----
  if (tid < 64) {
    float acc = xm_b2[0];
    for (int c = 0; c < 128; ++c)
      acc += (float)sm.scr.th.hid[tid * 128 + c] * xm_w2[c];
    sm.vx[tid] = acc;
  }
  __syncthreads();

  // ---- P = q @ R^T  (Toeplitz form of q.E), fp32 into scratch ----
  {
    int mt = (wv & 3) << 4;
    int mb = mt + ((lane >> 4) << 3);
    int nb = (wv >> 2) << 6;
#pragma unroll
    for (int i = 0; i < 4; ++i) {
      int nt = nb + (i << 4);
      v8f c = {};
      gemm_acc(c, sm.q_s, 128, Rbf, 128, mt, nt, 128, lane);
      int nn = nt + (lane & 15);
#pragma unroll
      for (int r = 0; r < 8; ++r) sm.scr.P[(mb + r) * 128 + nn] = c[r];
    }
  }
  __syncthreads();

  // ---- scores[t][s] = (q @ k^T)[t][s] + P[t][t-s+63] ----
  {
    int mt = (wv & 3) << 4;
    int mb = mt + ((lane >> 4) << 3);
    int nt0 = (wv >> 2) << 5;
#pragma unroll
    for (int i = 0; i < 2; ++i) {
      int nt = nt0 + (i << 4);
      v8f c = {};
      gemm_acc(c, sm.q_s, 128, sm.k_s, 128, mt, nt, 128, lane);
      int s = nt + (lane & 15);
#pragma unroll
      for (int r = 0; r < 8; ++r) {
        int t = mb + r;
        sm.sc[t * 64 + s] = c[r] + sm.scr.P[t * 128 + (t - s + 63)];
      }
    }
  }
  __syncthreads();

  // ---- softmax rows, alpha (bf16), shifted alpha, w = alpha*vx, rowsum ----
  if (tid < 64) {
    int t = tid;
    float mx = -3.0e38f;
    for (int s = 0; s < 64; ++s) mx = fmaxf(mx, sm.sc[t * 64 + s]);
    float sum = 0.f;
    for (int s = 0; s < 64; ++s) sum += __expf(sm.sc[t * 64 + s] - mx);
    float inv = 1.0f / sum, rs = 0.f;
    for (int s = 0; s < 64; ++s) {
      float a = __expf(sm.sc[t * 64 + s] - mx) * inv;
      sm.alpha_s[t * 64 + s] = (__bf16)a;
      float wval = a * sm.vx[s];
      sm.sc[t * 64 + s] = wval;          // overwrite scores with w
      rs += wval;
    }
    sm.rowsum[t] = rs;
    for (int r = 0; r < 128; ++r) {      // aShift[t][r] = alpha[t][t+63-r]
      int s = t + 63 - r;
      float a = (s >= 0 && s < 64) ? (float)sm.alpha_s[t * 64 + s] : 0.0f;
      sm.aSh_s[t * 128 + r] = (__bf16)a;
    }
  }
  __syncthreads();

  // ---- dh = alpha @ v + aShift @ R  ->  h_out = h + dh^T (vectorized) ----
  {
    float* hOutN = h_out + (size_t)n * H * T;
    int mt = (wv & 3) << 4;
    int mb = mt + ((lane >> 4) << 3);
    int nb = (wv >> 2) << 6;
#pragma unroll
    for (int i = 0; i < 4; ++i) {
      int nt = nb + (i << 4);
      v8f c = {};
      gemm_acc(c, sm.alpha_s, 64, sm.vT_s, 64, mt, nt, 64, lane);  // alpha @ v
      gemm_acc(c, sm.aSh_s, 128, Rt, 128, mt, nt, 128, lane);      // aShift @ R
      int ch = nt + (lane & 15);
      // c[r] corresponds to t = mb + r: contiguous in h[n][ch][t]
      v8f hv = *(const v8f*)(hN + ch * 64 + mb);
      *(v8f*)(hOutN + ch * 64 + mb) = hv + c;
    }
  }

  // ---- x update: x_out[d][t] = x[d][t]*(1+rowsum[t]) - sum_s w[t][s]*x[d][s]
  if (tid < 192) {
    int d = tid >> 6, t = tid & 63;
    float acc = 0.f;
    for (int s = 0; s < 64; ++s) acc += sm.sc[t * 64 + s] * sm.x_s[d * 64 + s];
    x_out[(size_t)n * 192 + d * 64 + t] =
        sm.x_s[d * 64 + t] * (1.0f + sm.rowsum[t]) - acc;
  }
}

// ---------------------------------------------------------------------------
extern "C" void kernel_launch(void* const* d_in, const int* in_sizes, int n_in,
                              void* d_out, int out_size, void* d_ws, size_t ws_size,
                              hipStream_t stream) {
  (void)in_sizes; (void)n_in; (void)out_size; (void)ws_size;
  const float* x     = (const float*)d_in[0];
  const float* h     = (const float*)d_in[1];
  const float* q_w1  = (const float*)d_in[2];
  const float* q_b1  = (const float*)d_in[3];
  const float* q_w2  = (const float*)d_in[4];
  const float* q_b2  = (const float*)d_in[5];
  const float* k_w1  = (const float*)d_in[6];
  const float* k_b1  = (const float*)d_in[7];
  const float* k_w2  = (const float*)d_in[8];
  const float* k_b2  = (const float*)d_in[9];
  const float* v_w1  = (const float*)d_in[10];
  const float* v_b1  = (const float*)d_in[11];
  const float* v_w2  = (const float*)d_in[12];
  const float* v_b2  = (const float*)d_in[13];
  const float* xm_w1 = (const float*)d_in[14];
  const float* xm_b1 = (const float*)d_in[15];
  const float* xm_w2 = (const float*)d_in[16];
  const float* xm_b2 = (const float*)d_in[17];
  const float* t_w   = (const float*)d_in[18];
  const float* t_b   = (const float*)d_in[19];

  __bf16* wsb = (__bf16*)d_ws;
  __bf16* Wq1  = wsb + 0 * 16384;
  __bf16* Wq2  = wsb + 1 * 16384;
  __bf16* Wk1  = wsb + 2 * 16384;
  __bf16* Wk2  = wsb + 3 * 16384;
  __bf16* Wv1  = wsb + 4 * 16384;
  __bf16* Wv2  = wsb + 5 * 16384;
  __bf16* Wxm1 = wsb + 6 * 16384;
  __bf16* Rbf  = wsb + 7 * 16384;
  __bf16* Rt   = wsb + 8 * 16384;

  prep_weights<<<(7 * 16384 + 255) / 256, 256, 0, stream>>>(
      q_w1, q_w2, k_w1, k_w2, v_w1, v_w2, xm_w1, wsb);
  prep_R<<<128, 128, 0, stream>>>(t_w, t_b, Rbf, Rt);

  float* x_out = (float*)d_out;
  float* h_out = x_out + (size_t)BN * 3 * T;
  et_layer_main<<<BN, 256, 0, stream>>>(
      x, h, q_b1, q_b2, k_b1, k_b2, v_b1, v_b2,
      xm_b1, xm_w2, xm_b2,
      Wq1, Wq2, Wk1, Wk2, Wv1, Wv2, Wxm1, Rbf, Rt,
      x_out, h_out);
}
